// DCE_74191265071716
// MI455X (gfx1250) — matrix operand order repaired and checked by reading, not compiled, not run on hardware
//
#include <hip/hip_runtime.h>
#include <math.h>

// ---------------------------------------------------------------------------
// DCE loss on gfx1250 via bf16 hi/lo-split WMMA (3 products ~ f32 accuracy).
//   dist^2 = |x|^2 + |p|^2 - 2 x.p ; loss[b] = cnt_b*log(sum_p e^{-d}) + sum_mask d
// d_in[0]=feature f32[B,128] d_in[1]=label i32[B]
// d_in[2]=proto  f32[P,128]  d_in[3]=proto_labels i32[P]
// d_out = f32[B]
// ---------------------------------------------------------------------------

typedef __attribute__((ext_vector_type(16))) __bf16 v16bf;
typedef __attribute__((ext_vector_type(8)))  float  v8f;

#define DIMK   128
#define WAVES  8
#define MT     16              // rows (B) per block
#define PT     (WAVES * 16)    // prototype columns per block iteration

__device__ __forceinline__ unsigned short bf16_trunc_bits(float x) {
    return (unsigned short)(__float_as_uint(x) >> 16);
}

// Split f32 rows into bf16 hi/lo planes (stored as u16) and compute row norms.
__global__ void dce_prep(const float* __restrict__ src,
                         unsigned short* __restrict__ hi,
                         unsigned short* __restrict__ lo,
                         float* __restrict__ n2, int rows) {
    int r = blockIdx.x * blockDim.x + threadIdx.x;
    if (r >= rows) return;
    const float* s = src + (size_t)r * DIMK;
    unsigned short* h = hi + (size_t)r * DIMK;
    unsigned short* l = lo + (size_t)r * DIMK;
    float acc = 0.0f;
#pragma unroll 8
    for (int k = 0; k < DIMK; ++k) {
        float x  = s[k];
        acc      = fmaf(x, x, acc);
        unsigned uh = __float_as_uint(x) & 0xFFFF0000u;
        float xh = __uint_as_float(uh);
        h[k] = (unsigned short)(uh >> 16);
        l[k] = bf16_trunc_bits(x - xh);
    }
    n2[r] = acc;
}

__global__ __launch_bounds__(256, 2)
void dce_main(const unsigned short* __restrict__ fh, const unsigned short* __restrict__ fl,
              const float* __restrict__ x2g, const int* __restrict__ label,
              const unsigned short* __restrict__ ph, const unsigned short* __restrict__ pl,
              const float* __restrict__ p2g, const int* __restrict__ plab,
              float* __restrict__ out, int P, int B) {
    __shared__ float sE[WAVES][MT];
    __shared__ float sD[WAVES][MT];
    __shared__ float sC[WAVES][MT];

    const int tid     = threadIdx.x;
    const int lane    = tid & 31;
    const int wave    = tid >> 5;
    const int l15     = lane & 15;
    const int hiHalf  = (lane >> 4) & 1;       // which K-half this lane holds
    const int rowBase = blockIdx.x * MT;

    union P16 { v16bf v; uint4 q[2]; };

    // ---- Preload A operands: 16 feature rows x 128 K, bf16 hi/lo, contiguous K order (same as B).
    // lane<16: K {kb+0..15}; lane>=16: K {kb+16..31}
    P16 Ah[4], Al[4];
    {
        int r = rowBase + l15; if (r >= B) r = B - 1;
        const size_t rowOff = (size_t)r * DIMK;
#pragma unroll
        for (int i = 0; i < 4; ++i) {
            const int k0 = 32 * i + (hiHalf ? 16 : 0);
            Ah[i].q[0] = *(const uint4*)(fh + rowOff + k0);
            Ah[i].q[1] = *(const uint4*)(fh + rowOff + k0 + 8);
            Al[i].q[0] = *(const uint4*)(fl + rowOff + k0);
            Al[i].q[1] = *(const uint4*)(fl + rowOff + k0 + 8);
        }
    }

    // Per-lane row constants: this lane's C-tile rows are hiHalf*8 + v.
    float xr[8]; int lr[8];
#pragma unroll
    for (int v = 0; v < 8; ++v) {
        int r = rowBase + hiHalf * 8 + v; if (r >= B) r = B - 1;
        xr[v] = x2g[r];
        lr[v] = label[r];
    }

    float sumE[8], sumD[8], cnt[8];
#pragma unroll
    for (int v = 0; v < 8; ++v) { sumE[v] = 0.f; sumD[v] = 0.f; cnt[v] = 0.f; }

    // ---- Main loop: only full PT-wide tiles (no masking, no divergence).
    const int fullIters = P / PT;
    const int cBase     = wave * 16 + l15;
    const size_t stepB  = (size_t)PT * DIMK;
    const unsigned short* pbh = ph + (size_t)cBase * DIMK + (hiHalf ? 16 : 0);
    const unsigned short* pbl = pl + (size_t)cBase * DIMK + (hiHalf ? 16 : 0);
    const float* p2p = p2g + cBase;
    const int*   plp = plab + cBase;

    for (int it = 0; it < fullIters; ++it) {
        const float p2c = *p2p;
        const int   pc  = *plp;
        if (it + 1 < fullIters) {                  // global_prefetch_b8
            __builtin_prefetch(pbh + stepB, 0, 1);
            __builtin_prefetch(pbl + stepB, 0, 1);
        }

        // Load the full 16x128 bf16 hi/lo B tile for this wave's 16 protos.
        P16 Bh[4], Bl[4];
#pragma unroll
        for (int i = 0; i < 4; ++i) {
            Bh[i].q[0] = *(const uint4*)(pbh + 32 * i);
            Bh[i].q[1] = *(const uint4*)(pbh + 32 * i + 8);
            Bl[i].q[0] = *(const uint4*)(pbl + 32 * i);
            Bl[i].q[1] = *(const uint4*)(pbl + 32 * i + 8);
        }

        v8f acc = {};
#pragma unroll
        for (int i = 0; i < 4; ++i) {
            // x.p ~= hi*hi + hi*lo + lo*hi  (f32 accumulate)
            acc = __builtin_amdgcn_wmma_f32_16x16x32_bf16(false, Ah[i].v, false, Bh[i].v,
                                                          (short)0, acc, false, false);
            acc = __builtin_amdgcn_wmma_f32_16x16x32_bf16(false, Ah[i].v, false, Bl[i].v,
                                                          (short)0, acc, false, false);
            acc = __builtin_amdgcn_wmma_f32_16x16x32_bf16(false, Al[i].v, false, Bh[i].v,
                                                          (short)0, acc, false, false);
        }

        // Branchless epilogue: lane holds column c, rows hiHalf*8+v in acc[v].
#pragma unroll
        for (int v = 0; v < 8; ++v) {
            float d2 = fmaf(-2.0f, acc[v], xr[v] + p2c);
            d2 = fmaxf(d2, 1e-12f);
            float d = sqrtf(d2);
            sumE[v] += __expf(-d);
            float mm = (pc == lr[v]) ? 1.0f : 0.0f;   // v_cmp + v_cndmask, no exec games
            sumD[v] = fmaf(mm, d, sumD[v]);
            cnt[v] += mm;
        }

        pbh += stepB; pbl += stepB; p2p += PT; plp += PT;
    }

    // ---- Tail: at most one partial tile, masked branchlessly.
    if (fullIters * PT < P) {
        const int  craw  = fullIters * PT + cBase;
        const bool valid = craw < P;
        const int  c     = valid ? craw : (P - 1);
        const size_t cOff = (size_t)c * DIMK + (hiHalf ? 16 : 0);
        const float p2c = p2g[c];
        const int   pc  = plab[c];

        P16 Bh[4], Bl[4];
#pragma unroll
        for (int i = 0; i < 4; ++i) {
            Bh[i].q[0] = *(const uint4*)(ph + cOff + 32 * i);
            Bh[i].q[1] = *(const uint4*)(ph + cOff + 32 * i + 8);
            Bl[i].q[0] = *(const uint4*)(pl + cOff + 32 * i);
            Bl[i].q[1] = *(const uint4*)(pl + cOff + 32 * i + 8);
        }
        v8f acc = {};
#pragma unroll
        for (int i = 0; i < 4; ++i) {
            acc = __builtin_amdgcn_wmma_f32_16x16x32_bf16(false, Ah[i].v, false, Bh[i].v,
                                                          (short)0, acc, false, false);
            acc = __builtin_amdgcn_wmma_f32_16x16x32_bf16(false, Ah[i].v, false, Bl[i].v,
                                                          (short)0, acc, false, false);
            acc = __builtin_amdgcn_wmma_f32_16x16x32_bf16(false, Al[i].v, false, Bh[i].v,
                                                          (short)0, acc, false, false);
        }
        const float vm = valid ? 1.0f : 0.0f;
#pragma unroll
        for (int v = 0; v < 8; ++v) {
            float d2 = fmaf(-2.0f, acc[v], xr[v] + p2c);
            d2 = fmaxf(d2, 1e-12f);
            float d = sqrtf(d2);
            sumE[v] = fmaf(vm, __expf(-d), sumE[v]);
            float mm = (valid && pc == lr[v]) ? 1.0f : 0.0f;
            sumD[v] = fmaf(mm, d, sumD[v]);
            cnt[v] += mm;
        }
    }

    // Reduce over the 16 columns held by each 16-lane half (deterministic).
#pragma unroll
    for (int v = 0; v < 8; ++v) {
#pragma unroll
        for (int m = 1; m < 16; m <<= 1) {
            sumE[v] += __shfl_xor(sumE[v], m, 32);
            sumD[v] += __shfl_xor(sumD[v], m, 32);
            cnt[v]  += __shfl_xor(cnt[v],  m, 32);
        }
    }
    if (l15 == 0) {
#pragma unroll
        for (int v = 0; v < 8; ++v) {
            const int r = hiHalf * 8 + v;
            sE[wave][r] = sumE[v]; sD[wave][r] = sumD[v]; sC[wave][r] = cnt[v];
        }
    }
    __syncthreads();

    if (tid < MT && rowBase + tid < B) {
        float E = 0.f, Dm = 0.f, Cn = 0.f;
#pragma unroll
        for (int w = 0; w < WAVES; ++w) { E += sE[w][tid]; Dm += sD[w][tid]; Cn += sC[w][tid]; }
        out[rowBase + tid] = Cn * logf(E) + Dm;
    }
}

extern "C" void kernel_launch(void* const* d_in, const int* in_sizes, int n_in,
                              void* d_out, int out_size, void* d_ws, size_t ws_size,
                              hipStream_t stream) {
    const float* feat  = (const float*)d_in[0];
    const int*   label = (const int*)  d_in[1];
    const float* prot  = (const float*)d_in[2];
    const int*   plab  = (const int*)  d_in[3];
    const int B = in_sizes[1];
    const int P = in_sizes[3];   // D is fixed at 128 (in_sizes[0] == B*128)

    // Workspace layout (256B-aligned slabs)
    char* ws = (char*)d_ws;
    size_t off = 0;
    auto take = [&](size_t bytes) { size_t o = off; off += (bytes + 255) & ~(size_t)255; return o; };
    float*          p2 = (float*)         (ws + take((size_t)P * 4));
    float*          x2 = (float*)         (ws + take((size_t)B * 4));
    unsigned short* ph = (unsigned short*)(ws + take((size_t)P * DIMK * 2));
    unsigned short* pl = (unsigned short*)(ws + take((size_t)P * DIMK * 2));
    unsigned short* fh = (unsigned short*)(ws + take((size_t)B * DIMK * 2));
    unsigned short* fl = (unsigned short*)(ws + take((size_t)B * DIMK * 2));
    (void)ws_size; (void)n_in; (void)out_size;

    dce_prep<<<(P + 255) / 256, 256, 0, stream>>>(prot, ph, pl, p2, P);
    dce_prep<<<(B + 255) / 256, 256, 0, stream>>>(feat, fh, fl, x2, B);

    dce_main<<<(B + MT - 1) / MT, 256, 0, stream>>>(fh, fl, x2, label,
                                                    ph, pl, p2, plab,
                                                    (float*)d_out, P, B);
}
